// Mamba2DecoderLayer_47347719471234
// MI455X (gfx1250) — compile-verified, matrix operand order checked by reading, not run on hardware
//
#include <hip/hip_runtime.h>
#include <stdint.h>

// ---------------------------------------------------------------- shapes
#define BSZ 2
#define TT 1024
#define HHID 2048
#define II 4096
#define NHEADS 64
#define HDIM 64
#define NSTATE 128
#define QCHUNK 128
#define NCHUNK (TT / QCHUNK)           // 8
#define KCONV 4
#define CONVD (II + 2 * NSTATE)        // 4352
#define DINP (2 * II + 2 * NSTATE + NHEADS) // 8512
#define BT (BSZ * TT)                  // 2048
#define EPSF 1e-5f

// ---------------------------------------------------------------- types
typedef __bf16 bf16_t;
typedef __attribute__((ext_vector_type(8)))  __bf16 v8bf;
typedef __attribute__((ext_vector_type(16))) __bf16 v16bf;
typedef __attribute__((ext_vector_type(8)))  float  v8f;
typedef __attribute__((ext_vector_type(4)))  int    v4i;

#define AS1 __attribute__((address_space(1)))
#define AS3 __attribute__((address_space(3)))

// ---------------------------------------------------------------- async copy
// gfx1250 async global->LDS copy (ASYNCcnt-tracked). Probe-confirmed signature:
// (AS1 v4i* gsrc, AS3 v4i* lds_dst, imm offset, imm cpol).
#if defined(__AMDGCN__) && __has_builtin(__builtin_amdgcn_global_load_async_to_lds_b128)
#define HAS_ASYNC 1
#define ASYNC_B128(dst, src)                                                   \
  __builtin_amdgcn_global_load_async_to_lds_b128((AS1 v4i*)(src),              \
                                                 (AS3 v4i*)(dst), 0, 0)
#else
#define HAS_ASYNC 0
#define ASYNC_B128(dst, src) (*(v8bf*)(dst) = *(const v8bf*)(src))
#endif

#if defined(__AMDGCN__) && __has_builtin(__builtin_amdgcn_s_wait_asynccnt)
#define WAIT_ASYNC() __builtin_amdgcn_s_wait_asynccnt(0)
#else
#define WAIT_ASYNC()
#endif

// ---------------------------------------------------------------- WMMA helpers
__device__ __forceinline__ v8f wmma_bf16(v16bf a, v16bf b, v8f c) {
#if defined(__AMDGCN__)
  return __builtin_amdgcn_wmma_f32_16x16x32_bf16(false, a, false, b, (short)0,
                                                 c, false, false);
#else
  return c;
#endif
}

// A/B fragment from LDS, tile stored row-major with K contiguous.
// 16-bit A layout (ISA 7.12.2): lane holds row = lane%16,
// element e holds K = 16*(e/8) + 8*(lane/16) + (e&7)  -> two b128 LDS loads.
__device__ __forceinline__ v16bf frag_rowk(const bf16_t* base, int rstride,
                                           int lane) {
  const int r = lane & 15, hh = (lane >> 4) & 1;
  const bf16_t* p = base + r * rstride + 8 * hh;
  v8bf lo = *(const v8bf*)(p);
  v8bf hi = *(const v8bf*)(p + 16);
  v16bf f;
#pragma unroll
  for (int e = 0; e < 8; ++e) { f[e] = lo[e]; f[e + 8] = hi[e]; }
  return f;
}

// B fragment when the tile is stored K-major (K x N row-major): lane holds
// column n = lane%16, elements walk K. Scalar LDS reads.
__device__ __forceinline__ v16bf frag_colk(const bf16_t* base, int rstride,
                                           int lane) {
  const int n = lane & 15, hh = (lane >> 4) & 1;
  v16bf f;
#pragma unroll
  for (int e = 0; e < 16; ++e) {
    int k = 16 * (e >> 3) + 8 * hh + (e & 7);
    f[e] = base[k * rstride + n];
  }
  return f;
}

__device__ __forceinline__ float siluf(float x) {
  return x / (1.f + __expf(-x));
}

// ---------------------------------------------------------------- kernels

// residual = hidden + residual ; x = rmsnorm(residual) -> bf16
__global__ __launch_bounds__(256) void k_addnorm(
    const float* __restrict__ hs, const float* __restrict__ res,
    const float* __restrict__ w, float* __restrict__ res_out,
    bf16_t* __restrict__ xbf) {
  const int row = blockIdx.x, tid = threadIdx.x;
  const float* hp = hs + (size_t)row * HHID;
  const float* rp = res + (size_t)row * HHID;
  float v[8], ss = 0.f;
#pragma unroll
  for (int i = 0; i < 8; ++i) {
    int c = tid + i * 256;
    float x = hp[c] + rp[c];
    v[i] = x;
    res_out[(size_t)row * HHID + c] = x;
    ss += x * x;
  }
  __shared__ float red[256];
  red[tid] = ss; __syncthreads();
  for (int s = 128; s > 0; s >>= 1) {
    if (tid < s) red[tid] += red[tid + s];
    __syncthreads();
  }
  float inv = rsqrtf(red[0] / (float)HHID + EPSF);
#pragma unroll
  for (int i = 0; i < 8; ++i) {
    int c = tid + i * 256;
    xbf[(size_t)row * HHID + c] = (bf16_t)(v[i] * inv * w[c]);
  }
}

__global__ __launch_bounds__(256) void k_cast_bf16(const float* __restrict__ in,
                                                   bf16_t* __restrict__ out,
                                                   int n) {
  int i = blockIdx.x * 256 + threadIdx.x;
  if (i < n) out[i] = (bf16_t)in[i];
}

// Generic C[M,N] = A[M,K] * B[N,K]^T, all bf16 in / f32 out.
// Block tile 128x64, 8 waves, each wave 16x64 (4 wmma accumulators).
#define GBM 128
#define GBN 64
#define GBK 32
#define APAD 40
__global__ __launch_bounds__(256) void k_gemm_abT(
    const bf16_t* __restrict__ A, const bf16_t* __restrict__ B,
    float* __restrict__ C, int M, int N, int Kd) {
  __shared__ __align__(16) bf16_t sA[GBM * APAD];
  __shared__ __align__(16) bf16_t sB[GBN * APAD];
  const int tid = threadIdx.x, wave = tid >> 5, lane = tid & 31;
  const int m0 = blockIdx.x * GBM, n0 = blockIdx.y * GBN;
  v8f acc[4] = {};
  for (int k0 = 0; k0 < Kd; k0 += GBK) {
#pragma unroll
    for (int i = 0; i < 2; ++i) {   // A: 128x32 = 512 x 8-elt chunks
      int ch = tid + i * 256;
      int r = ch >> 2, c8 = (ch & 3) * 8;
      ASYNC_B128(&sA[r * APAD + c8], &A[(size_t)(m0 + r) * Kd + k0 + c8]);
    }
    {                                // B: 64x32 = 256 x 8-elt chunks
      int r = tid >> 2, c8 = (tid & 3) * 8;
      ASYNC_B128(&sB[r * APAD + c8], &B[(size_t)(n0 + r) * Kd + k0 + c8]);
    }
    if (k0 + GBK < Kd)
      __builtin_prefetch(&A[(size_t)(m0 + (tid >> 1)) * Kd + k0 + GBK], 0, 1);
    WAIT_ASYNC();
    __syncthreads();
    v16bf af = frag_rowk(&sA[(wave * 16) * APAD], APAD, lane);
#pragma unroll
    for (int j = 0; j < 4; ++j) {
      v16bf bfr = frag_rowk(&sB[(j * 16) * APAD], APAD, lane);
      acc[j] = wmma_bf16(af, bfr, acc[j]);
    }
    __syncthreads();
  }
  const int mr = 8 * (lane >> 4), nc = lane & 15;
#pragma unroll
  for (int j = 0; j < 4; ++j)
#pragma unroll
    for (int e = 0; e < 8; ++e) {
      int m = m0 + wave * 16 + mr + e;
      int n = n0 + j * 16 + nc;
      C[(size_t)m * N + n] = acc[j][e];
    }
}

// causal depthwise conv (K=4) + SiLU over proj[:, I : I+CONVD]
__global__ __launch_bounds__(256) void k_conv(const float* __restrict__ proj,
                                              const float* __restrict__ cw,
                                              const float* __restrict__ cb,
                                              float* __restrict__ convo) {
  int idx = blockIdx.x * 256 + threadIdx.x;
  if (idx >= BT * CONVD) return;
  int c = idx % CONVD, bt = idx / CONVD;
  int b = bt / TT, t = bt % TT;
  float acc = cb[c];
#pragma unroll
  for (int k = 0; k < KCONV; ++k) {
    int t2 = t + k - (KCONV - 1);
    if (t2 >= 0)
      acc += proj[(size_t)(b * TT + t2) * DINP + II + c] * cw[c * KCONV + k];
  }
  convo[idx] = siluf(acc);
}

// dt = softplus(proj_dt + bias); a = dt * (-exp(A_log)) stored [b,h,c,l]
__global__ __launch_bounds__(256) void k_dt(const float* __restrict__ proj,
                                            const float* __restrict__ dtb,
                                            const float* __restrict__ alog,
                                            float* __restrict__ dt_out,
                                            float* __restrict__ a_out) {
  int idx = blockIdx.x * 256 + threadIdx.x;
  if (idx >= BT * NHEADS) return;
  int hh = idx % NHEADS, bt = idx / NHEADS;
  float x = proj[(size_t)bt * DINP + II + CONVD + hh] + dtb[hh];
  float dt = (x > 20.f) ? x : log1pf(__expf(x));
  dt_out[idx] = dt;
  int b = bt / TT, t = bt % TT, c = t / QCHUNK, l = t % QCHUNK;
  a_out[((size_t)(b * NHEADS + hh) * NCHUNK + c) * QCHUNK + l] =
      dt * (-__expf(alog[hh]));
}

// inclusive cumsum of a over l within a chunk
__global__ __launch_bounds__(128) void k_cumsum(const float* __restrict__ a,
                                                float* __restrict__ acum) {
  int g = blockIdx.x, tid = threadIdx.x;
  __shared__ float sc[QCHUNK];
  sc[tid] = a[(size_t)g * QCHUNK + tid];
  __syncthreads();
  for (int off = 1; off < QCHUNK; off <<= 1) {
    float v = (tid >= off) ? sc[tid - off] : 0.f;
    __syncthreads();
    sc[tid] += v;
    __syncthreads();
  }
  acum[(size_t)g * QCHUNK + tid] = sc[tid];
}

// X' = xs * dt  (bf16, [b,t,h,p])
__global__ __launch_bounds__(256) void k_xprime(const float* __restrict__ convo,
                                                const float* __restrict__ dt,
                                                bf16_t* __restrict__ xp) {
  int idx = blockIdx.x * 256 + threadIdx.x;
  if (idx >= BT * II) return;
  int bt = idx / II, i = idx % II, hh = i / HDIM;
  xp[idx] = (bf16_t)(convo[(size_t)bt * CONVD + i] * dt[bt * NHEADS + hh]);
}

// cast conv B/C blocks to bf16 [b,t,n]
__global__ __launch_bounds__(256) void k_castBC(const float* __restrict__ convo,
                                                bf16_t* __restrict__ Bc,
                                                bf16_t* __restrict__ Cc) {
  int idx = blockIdx.x * 256 + threadIdx.x;
  if (idx >= BT * NSTATE) return;
  int bt = idx / NSTATE, n = idx % NSTATE;
  Bc[idx] = (bf16_t)convo[(size_t)bt * CONVD + II + n];
  Cc[idx] = (bf16_t)convo[(size_t)bt * CONVD + II + NSTATE + n];
}

// Bdecay[b,c,h,s,n] = B[b,t,n] * exp(acum_last - acum[s])
__global__ __launch_bounds__(256) void k_bdecay(const float* __restrict__ convo,
                                                const float* __restrict__ acum,
                                                bf16_t* __restrict__ bd) {
  int idx = blockIdx.x * 256 + threadIdx.x;
  if (idx >= BSZ * NCHUNK * NHEADS * QCHUNK * NSTATE) return;
  int n = idx % NSTATE, r = idx / NSTATE;
  int s = r % QCHUNK; r /= QCHUNK;
  int hh = r % NHEADS; r /= NHEADS;
  int c = r % NCHUNK, b = r / NCHUNK;
  const float* ac = acum + ((size_t)(b * NHEADS + hh) * NCHUNK + c) * QCHUNK;
  float dec = __expf(ac[QCHUNK - 1] - ac[s]);
  bd[idx] =
      (bf16_t)(convo[(size_t)(b * TT + c * QCHUNK + s) * CONVD + II + n] * dec);
}

// Y_diag: per (b,c,h): scores = Cc*Bc^T (WMMA), * L mask/decay, then @ X'
#define SPAD 136
#define KPAD 40
__global__ __launch_bounds__(256) void k_ssd_diag(
    const bf16_t* __restrict__ Cc, const bf16_t* __restrict__ Bc,
    const bf16_t* __restrict__ xp, const float* __restrict__ acum,
    float* __restrict__ y) {
  const int hh = blockIdx.x, c = blockIdx.y, b = blockIdx.z;
  const int tid = threadIdx.x, wave = tid >> 5, lane = tid & 31;
  __shared__ __align__(16) bf16_t sS[QCHUNK * SPAD];  // scores (bf16)
  __shared__ __align__(16) bf16_t sCk[QCHUNK * KPAD]; // 128 x 32 K-slice of C
  __shared__ __align__(16) bf16_t sBk[QCHUNK * KPAD]; // 128 x 32 K-slice of B
  __shared__ float sAc[QCHUNK];
  const size_t rowbase = (size_t)(b * TT + c * QCHUNK);
  if (tid < QCHUNK)
    sAc[tid] = acum[((size_t)(b * NHEADS + hh) * NCHUNK + c) * QCHUNK + tid];

  // ---- stage 1: scores = Cc (Q x N) * Bc^T, streamed over K = NSTATE
  v8f acc[8] = {};
  for (int k0 = 0; k0 < NSTATE; k0 += 32) {
#pragma unroll
    for (int i = 0; i < 2; ++i) {
      int ch = tid + i * 256;
      int r = ch >> 2, c8 = (ch & 3) * 8;
      ASYNC_B128(&sCk[r * KPAD + c8], &Cc[(rowbase + r) * NSTATE + k0 + c8]);
      ASYNC_B128(&sBk[r * KPAD + c8], &Bc[(rowbase + r) * NSTATE + k0 + c8]);
    }
    WAIT_ASYNC();
    __syncthreads();
    v16bf af = frag_rowk(&sCk[(wave * 16) * KPAD], KPAD, lane);
#pragma unroll
    for (int j = 0; j < 8; ++j) {
      v16bf bfr = frag_rowk(&sBk[(j * 16) * KPAD], KPAD, lane);
      acc[j] = wmma_bf16(af, bfr, acc[j]);
    }
    __syncthreads();
  }
  // apply causal mask + segment decay, stash as bf16 A-operand for stage 2
  const int mr = 8 * (lane >> 4), nc = lane & 15;
#pragma unroll
  for (int j = 0; j < 8; ++j)
#pragma unroll
    for (int e = 0; e < 8; ++e) {
      int l = wave * 16 + mr + e, s = j * 16 + nc;
      float vv = (l >= s) ? acc[j][e] * __expf(sAc[l] - sAc[s]) : 0.f;
      sS[l * SPAD + s] = (bf16_t)vv;
    }
  __syncthreads();

  // ---- stage 2: Y = scores (Q x Q) @ X' (Q x P), stream 32-row X' slices
  v8f acc2[4] = {};
  for (int k0 = 0; k0 < QCHUNK; k0 += 32) {
    {
      int r = tid >> 3, c8 = (tid & 7) * 8; // 32 x 64 slice
      ASYNC_B128(&sBk[r * 72 + c8],
                 &xp[((rowbase + k0 + r) * NHEADS + hh) * HDIM + c8]);
    }
    WAIT_ASYNC();
    __syncthreads();
    v16bf af = frag_rowk(&sS[(wave * 16) * SPAD + k0], SPAD, lane);
#pragma unroll
    for (int j = 0; j < 4; ++j) {
      v16bf bfr = frag_colk(&sBk[j * 16], 72, lane);
      acc2[j] = wmma_bf16(af, bfr, acc2[j]);
    }
    __syncthreads();
  }
#pragma unroll
  for (int j = 0; j < 4; ++j)
#pragma unroll
    for (int e = 0; e < 8; ++e) {
      int l = wave * 16 + mr + e, p = j * 16 + nc;
      y[((rowbase + l) * NHEADS + hh) * HDIM + p] = acc2[j][e];
    }
}

// states[b,c,h,p,n] = sum_s X'[s,p] * Bdecay[s,n]   (transposed staging)
__global__ __launch_bounds__(128) void k_ssd_states(
    const bf16_t* __restrict__ bd, const bf16_t* __restrict__ xp,
    float* __restrict__ states) {
  const int hh = blockIdx.x, c = blockIdx.y, b = blockIdx.z;
  const int tid = threadIdx.x, wave = tid >> 5, lane = tid & 31;
  __shared__ __align__(16) bf16_t sXt[HDIM * SPAD];   // [p][s]
  __shared__ __align__(16) bf16_t sBt[NSTATE * SPAD]; // [n][s]
  const size_t rowbase = (size_t)(b * TT + c * QCHUNK);
  const size_t bdbase =
      ((size_t)((b * NCHUNK + c) * NHEADS) + hh) * (QCHUNK * NSTATE);
  for (int idx = tid; idx < QCHUNK * HDIM; idx += 128) {
    int s = idx / HDIM, p = idx % HDIM;
    sXt[p * SPAD + s] = xp[((rowbase + s) * NHEADS + hh) * HDIM + p];
  }
  for (int idx = tid; idx < QCHUNK * NSTATE; idx += 128) {
    int s = idx / NSTATE, n = idx % NSTATE;
    sBt[n * SPAD + s] = bd[bdbase + (size_t)s * NSTATE + n];
  }
  __syncthreads();
  v8f acc[8] = {};
  for (int k0 = 0; k0 < QCHUNK; k0 += 32) {
    v16bf af = frag_rowk(&sXt[(wave * 16) * SPAD + k0], SPAD, lane);
#pragma unroll
    for (int j = 0; j < 8; ++j) {
      v16bf bfr = frag_rowk(&sBt[(j * 16) * SPAD + k0], SPAD, lane);
      acc[j] = wmma_bf16(af, bfr, acc[j]);
    }
  }
  const int mr = 8 * (lane >> 4), nc = lane & 15;
  size_t sb = ((size_t)((b * NCHUNK + c) * NHEADS) + hh) * (HDIM * NSTATE);
#pragma unroll
  for (int j = 0; j < 8; ++j)
#pragma unroll
    for (int e = 0; e < 8; ++e) {
      int p = wave * 16 + mr + e, n = j * 16 + nc;
      states[sb + (size_t)p * NSTATE + n] = acc[j][e];
    }
}

// inter-chunk recurrence: prev[c] = exp(tot[c-1])*prev[c-1] + states[c-1]
__global__ __launch_bounds__(256) void k_scan(const float* __restrict__ states,
                                              const float* __restrict__ acum,
                                              float* __restrict__ prev) {
  int g = blockIdx.x, b = g / NHEADS, hh = g % NHEADS;
  int tid = threadIdx.x;
  float dec[NCHUNK];
#pragma unroll
  for (int c = 0; c < NCHUNK; ++c)
    dec[c] = __expf(
        acum[((size_t)(b * NHEADS + hh) * NCHUNK + c) * QCHUNK + QCHUNK - 1]);
  for (int e = tid; e < HDIM * NSTATE; e += 256) {
    float run = 0.f;
    for (int c = 0; c < NCHUNK; ++c) {
      size_t base =
          ((size_t)((b * NCHUNK + c) * NHEADS) + hh) * (HDIM * NSTATE) + e;
      prev[base] = run;
      run = run * dec[c] + states[base];
    }
  }
}

// Y_off: y += exp(acum[l]) * (Cc @ prev^T) + D*xs
__global__ __launch_bounds__(256) void k_ssd_off(
    const bf16_t* __restrict__ Cc, const float* __restrict__ prev,
    const float* __restrict__ acum, const float* __restrict__ convo,
    const float* __restrict__ Dw, float* __restrict__ y) {
  const int hh = blockIdx.x, c = blockIdx.y, b = blockIdx.z;
  const int tid = threadIdx.x, wave = tid >> 5, lane = tid & 31;
  __shared__ __align__(16) bf16_t sC[QCHUNK * SPAD];
  __shared__ __align__(16) bf16_t sP[HDIM * SPAD];
  __shared__ float sAc[QCHUNK];
  const size_t rowbase = (size_t)(b * TT + c * QCHUNK);
#pragma unroll
  for (int i = 0; i < 8; ++i) {
    int ch = tid + i * 256;
    int r = ch >> 4, c8 = (ch & 15) * 8;
    ASYNC_B128(&sC[r * SPAD + c8], &Cc[(rowbase + r) * NSTATE + c8]);
  }
  size_t pb = ((size_t)((b * NCHUNK + c) * NHEADS) + hh) * (HDIM * NSTATE);
  for (int idx = tid; idx < HDIM * NSTATE; idx += 256)
    sP[(idx / NSTATE) * SPAD + (idx % NSTATE)] = (bf16_t)prev[pb + idx];
  if (tid < QCHUNK)
    sAc[tid] = acum[((size_t)(b * NHEADS + hh) * NCHUNK + c) * QCHUNK + tid];
  WAIT_ASYNC();
  __syncthreads();
  v8f acc[4] = {};
  for (int k0 = 0; k0 < NSTATE; k0 += 32) {
    v16bf af = frag_rowk(&sC[(wave * 16) * SPAD + k0], SPAD, lane);
#pragma unroll
    for (int j = 0; j < 4; ++j) {
      v16bf bfr = frag_rowk(&sP[(j * 16) * SPAD + k0], SPAD, lane);
      acc[j] = wmma_bf16(af, bfr, acc[j]);
    }
  }
  const int mr = 8 * (lane >> 4), nc = lane & 15;
  float dh = Dw[hh];
#pragma unroll
  for (int j = 0; j < 4; ++j)
#pragma unroll
    for (int e = 0; e < 8; ++e) {
      int l = wave * 16 + mr + e, p = j * 16 + nc;
      size_t yi = ((rowbase + l) * NHEADS + hh) * HDIM + p;
      float xs = convo[(size_t)(rowbase + l) * CONVD + hh * HDIM + p];
      y[yi] = y[yi] + __expf(sAc[l]) * acc[j][e] + dh * xs;
    }
}

// y = rmsnorm(y * silu(gate)) -> bf16
__global__ __launch_bounds__(256) void k_gatednorm(
    const float* __restrict__ y, const float* __restrict__ proj,
    const float* __restrict__ gw, bf16_t* __restrict__ ybf) {
  const int row = blockIdx.x, tid = threadIdx.x;
  float v[16], ss = 0.f;
#pragma unroll
  for (int i = 0; i < 16; ++i) {
    int c = tid + i * 256;
    float g = proj[(size_t)row * DINP + c];
    float val = y[(size_t)row * II + c] * siluf(g);
    v[i] = val;
    ss += val * val;
  }
  __shared__ float red[256];
  red[tid] = ss; __syncthreads();
  for (int s = 128; s > 0; s >>= 1) {
    if (tid < s) red[tid] += red[tid + s];
    __syncthreads();
  }
  float inv = rsqrtf(red[0] / (float)II + EPSF);
#pragma unroll
  for (int i = 0; i < 16; ++i) {
    int c = tid + i * 256;
    ybf[(size_t)row * II + c] = (bf16_t)(v[i] * inv * gw[c]);
  }
}

// ---------------------------------------------------------------- workspace map
static constexpr size_t OFF_XBF  = 0;
static constexpr size_t OFF_WIBF = OFF_XBF  + (size_t)BT * HHID * 2;
static constexpr size_t OFF_WOBF = OFF_WIBF + (size_t)DINP * HHID * 2;
static constexpr size_t OFF_PROJ = OFF_WOBF + (size_t)HHID * II * 2;
static constexpr size_t OFF_CONV = OFF_PROJ + (size_t)BT * DINP * 4;
static constexpr size_t OFF_DT   = OFF_CONV + (size_t)BT * CONVD * 4;
static constexpr size_t OFF_ARAW = OFF_DT   + (size_t)BT * NHEADS * 4;
static constexpr size_t OFF_ACUM = OFF_ARAW + (size_t)BSZ * NHEADS * TT * 4;
static constexpr size_t OFF_XPBF = OFF_ACUM + (size_t)BSZ * NHEADS * TT * 4;
static constexpr size_t OFF_BCBF = OFF_XPBF + (size_t)BT * II * 2;
static constexpr size_t OFF_CCBF = OFF_BCBF + (size_t)BT * NSTATE * 2;
static constexpr size_t OFF_BDBF = OFF_CCBF + (size_t)BT * NSTATE * 2;
static constexpr size_t OFF_ST   = OFF_BDBF + (size_t)BSZ * NCHUNK * NHEADS * QCHUNK * NSTATE * 2;
static constexpr size_t OFF_PREV = OFF_ST   + (size_t)BSZ * NCHUNK * NHEADS * HDIM * NSTATE * 4;
static constexpr size_t OFF_Y    = OFF_PREV + (size_t)BSZ * NCHUNK * NHEADS * HDIM * NSTATE * 4;
static constexpr size_t OFF_YBF  = OFF_Y    + (size_t)BT * II * 4;

// ---------------------------------------------------------------- launch
extern "C" void kernel_launch(void* const* d_in, const int* in_sizes, int n_in,
                              void* d_out, int out_size, void* d_ws,
                              size_t ws_size, hipStream_t stream) {
  (void)in_sizes; (void)n_in; (void)out_size; (void)ws_size;
  const float* hs   = (const float*)d_in[0];
  const float* res  = (const float*)d_in[1];
  const float* nw   = (const float*)d_in[2];
  const float* wi   = (const float*)d_in[3];
  const float* cw   = (const float*)d_in[4];
  const float* cb   = (const float*)d_in[5];
  const float* alog = (const float*)d_in[6];
  const float* Dw   = (const float*)d_in[7];
  const float* dtb  = (const float*)d_in[8];
  const float* gnw  = (const float*)d_in[9];
  const float* wo   = (const float*)d_in[10];

  float* outp    = (float*)d_out;
  float* res_out = outp + (size_t)BT * HHID;

  char* ws = (char*)d_ws;
  bf16_t* xbf     = (bf16_t*)(ws + OFF_XBF);
  bf16_t* wibf    = (bf16_t*)(ws + OFF_WIBF);
  bf16_t* wobf    = (bf16_t*)(ws + OFF_WOBF);
  float*  projp   = (float*)(ws + OFF_PROJ);
  float*  convp   = (float*)(ws + OFF_CONV);
  float*  dtp     = (float*)(ws + OFF_DT);
  float*  arawp   = (float*)(ws + OFF_ARAW);
  float*  acump   = (float*)(ws + OFF_ACUM);
  bf16_t* xpbf    = (bf16_t*)(ws + OFF_XPBF);
  bf16_t* Bcbf    = (bf16_t*)(ws + OFF_BCBF);
  bf16_t* Ccbf    = (bf16_t*)(ws + OFF_CCBF);
  bf16_t* bdbf    = (bf16_t*)(ws + OFF_BDBF);
  float*  statesp = (float*)(ws + OFF_ST);
  float*  prevp   = (float*)(ws + OFF_PREV);
  float*  yp      = (float*)(ws + OFF_Y);
  bf16_t* ybf     = (bf16_t*)(ws + OFF_YBF);

  k_addnorm<<<BT, 256, 0, stream>>>(hs, res, nw, res_out, xbf);
  k_cast_bf16<<<(DINP * HHID + 255) / 256, 256, 0, stream>>>(wi, wibf,
                                                             DINP * HHID);
  k_cast_bf16<<<(HHID * II + 255) / 256, 256, 0, stream>>>(wo, wobf,
                                                           HHID * II);
  k_gemm_abT<<<dim3(BT / GBM, DINP / GBN), 256, 0, stream>>>(
      xbf, wibf, projp, BT, DINP, HHID);
  k_conv<<<(BT * CONVD + 255) / 256, 256, 0, stream>>>(projp, cw, cb, convp);
  k_dt<<<(BT * NHEADS + 255) / 256, 256, 0, stream>>>(projp, dtb, alog, dtp,
                                                      arawp);
  k_cumsum<<<BSZ * NHEADS * NCHUNK, 128, 0, stream>>>(arawp, acump);
  k_xprime<<<(BT * II + 255) / 256, 256, 0, stream>>>(convp, dtp, xpbf);
  k_castBC<<<(BT * NSTATE + 255) / 256, 256, 0, stream>>>(convp, Bcbf, Ccbf);
  k_bdecay<<<(BSZ * NCHUNK * NHEADS * QCHUNK * NSTATE + 255) / 256, 256, 0,
             stream>>>(convp, acump, bdbf);
  dim3 gssd(NHEADS, NCHUNK, BSZ);
  k_ssd_diag<<<gssd, 256, 0, stream>>>(Ccbf, Bcbf, xpbf, acump, yp);
  k_ssd_states<<<gssd, 128, 0, stream>>>(bdbf, xpbf, statesp);
  k_scan<<<BSZ * NHEADS, 256, 0, stream>>>(statesp, acump, prevp);
  k_ssd_off<<<gssd, 256, 0, stream>>>(Ccbf, prevp, acump, convp, Dw, yp);
  k_gatednorm<<<BT, 256, 0, stream>>>(yp, projp, gnw, ybf);
  k_gemm_abT<<<dim3(BT / GBM, HHID / GBN), 256, 0, stream>>>(
      ybf, wobf, outp, BT, HHID, II);
}
